// TransformerModel_43404939493640
// MI455X (gfx1250) — compile-verified, matrix operand order checked by reading, not run on hardware
//
#include <hip/hip_runtime.h>

// ---------------------------------------------------------------------------
// Transformer forward for MI455X (gfx1250), bf16 WMMA everywhere.
//   N_LAYER=4, N_HEAD=8, N_DICT=32000, D_MODEL=256, D_K=32, D_HIDDEN=1024,
//   B=2, T=2048  -> B*T = 4096 rows.
// GEMMs stage A/B K-panels into LDS with gfx1250 async copies
// (global_load_async_to_lds_b128 + s_wait_asynccnt), double buffered, and
// feed WMMA operands from ds_load_b128. Attention is a single-pass flash
// loop (reference softmax has no max-subtraction) with WMMA row-sums.
// Assumes ws_size >= ~96 MB.
// ---------------------------------------------------------------------------

typedef __attribute__((ext_vector_type(16))) __bf16 v16bf;
typedef __attribute__((ext_vector_type(8)))  __bf16 v8bf;
typedef __attribute__((ext_vector_type(8)))  float  v8f;
typedef __attribute__((ext_vector_type(4)))  int    v4i;

#define N_LAYER  4
#define N_HEAD   8
#define N_DICT   32000
#define D_MODEL  256
#define D_K      32
#define D_HIDDEN 1024
#define BB       2
#define TT       2048
#define BT       (BB * TT)   // 4096

// ---- WMMA helpers ----------------------------------------------------------

__device__ __forceinline__ v8f zero8() {
  v8f z;
#pragma unroll
  for (int i = 0; i < 8; ++i) z[i] = 0.0f;
  return z;
}

// Operand loader for v_wmma_f32_16x16x32_bf16.
// Per ISA layout (16-bit A 16x32): lanes 0-15 hold K = {0..7, 16..23},
// lanes 16-31 hold K = {8..15, 24..31}. Caller passes p = rowBase + kh*8,
// we fetch two contiguous 16-byte chunks at p and p+16.
__device__ __forceinline__ v16bf load_op(const __bf16* p) {
  v8bf lo = *(const v8bf*)(p);
  v8bf hi = *(const v8bf*)(p + 16);
  v16bf r;
#pragma unroll
  for (int i = 0; i < 8; ++i) { r[i] = lo[i]; r[i + 8] = hi[i]; }
  return r;
}

__device__ __forceinline__ v8f wmma_bf16(v16bf a, v16bf b, v8f c) {
  return __builtin_amdgcn_wmma_f32_16x16x32_bf16(
      /*neg_a=*/false, a, /*neg_b=*/false, b,
      /*c_mod=*/(short)0, c, /*reuse_a=*/false, /*reuse_b=*/false);
}

// ---- gfx1250 async global->LDS copy (ASYNCcnt path) ------------------------

typedef __attribute__((address_space(1))) v4i* gv4i_p;   // global 16B vector
typedef __attribute__((address_space(3))) v4i* lv4i_p;   // LDS 16B vector

__device__ __forceinline__ void async_cp16(void* lds, const void* glb) {
#if __has_builtin(__builtin_amdgcn_global_load_async_to_lds_b128)
  __builtin_amdgcn_global_load_async_to_lds_b128(
      (gv4i_p)(unsigned long long)glb,
      (lv4i_p)(unsigned)(unsigned long long)lds,   // generic->LDS: low 32 bits
      0, 0);
#else
  unsigned           l = (unsigned)(unsigned long long)lds;
  unsigned long long g = (unsigned long long)glb;
  asm volatile("global_load_async_to_lds_b128 %0, %1, off"
               :: "v"(l), "v"(g) : "memory");
#endif
}

__device__ __forceinline__ void wait_async0() {
#if __has_builtin(__builtin_amdgcn_s_wait_asynccnt)
  __builtin_amdgcn_s_wait_asynccnt(0);
#else
  asm volatile("s_wait_asynccnt 0x0" ::: "memory");
#endif
}

// ---- Generic tiled GEMM:  C[M,N] = A[M,K](row-major bf16) x Bt[N,K]^T ------
// 256 threads = 8 waves. Block tile = 128 rows x 64 cols; wave w owns rows
// [w*16, w*16+16). Per 32-wide K step the block async-copies an 8 KB A panel
// (128x32) and a 4 KB B panel (64x32) into LDS (3 b128 copies per thread),
// double buffered; WMMA operands are then ds_load_b128 reads.
// grid = (N/64, M/128, Z).  MODE: 0=f32 store, 1=bf16 store, 2=bf16 relu,
// 3=residual f32 add + bf16 mirror, 4=f32 + bias.
template <int MODE>
__global__ __launch_bounds__(256) void gemm_k(
    const __bf16* __restrict__ A,  int lda, long long aSz,
    const __bf16* __restrict__ Bt, int ldb, long long bSz,
    void* __restrict__ Cp,         int ldc, long long cSz,
    int K,
    float* __restrict__ residF, __bf16* __restrict__ residBf,
    const float* __restrict__ bias)
{
  __shared__ __bf16 ldsA[2][128][32];   // 16 KB
  __shared__ __bf16 ldsB[2][64][32];    //  8 KB

  const int tid  = threadIdx.x;
  const int lane = tid & 31;
  const int wave = tid >> 5;
  const int ml   = lane & 15;     // row (A) / col (B) within 16x16 tile
  const int kh   = lane >> 4;     // K half-group select
  const int mBlk = blockIdx.y * 128;
  const int n0   = blockIdx.x * 64;
  const long long z = blockIdx.z;

  A  += z * aSz;
  Bt += z * bSz;

  // Stage K-panel k0 into LDS buffer bi: A 512 chunks (2/thread), B 256 (1/thread).
  auto stage = [&](int k0, int bi) {
#pragma unroll
    for (int i = 0; i < 2; ++i) {
      const int c = tid + 256 * i;            // chunk id: 4 chunks per 32-col row
      const int row = c >> 2, coff = (c & 3) * 8;
      async_cp16(&ldsA[bi][row][coff],
                 A + (long long)(mBlk + row) * lda + k0 + coff);
    }
    {
      const int row = tid >> 2, coff = (tid & 3) * 8;
      async_cp16(&ldsB[bi][row][coff],
                 Bt + (long long)(n0 + row) * ldb + k0 + coff);
    }
  };

  v8f acc[4];
#pragma unroll
  for (int t = 0; t < 4; ++t) acc[t] = zero8();

  const int nk = K >> 5;
  stage(0, 0);
  for (int ks = 0; ks < nk; ++ks) {
    wait_async0();        // this wave's copies for step ks are in LDS
    __syncthreads();      // -> every wave's copies are; prev buffer fully read
    if (ks + 1 < nk) stage((ks + 1) << 5, (ks + 1) & 1);

    const int bi = ks & 1;
    v16bf a = load_op(&ldsA[bi][wave * 16 + ml][kh * 8]);
#pragma unroll
    for (int t = 0; t < 4; ++t) {
      v16bf b = load_op(&ldsB[bi][t * 16 + ml][kh * 8]);
      acc[t] = wmma_bf16(a, b, acc[t]);
    }
  }

  // C layout: lane holds (m = m0 + kh*8 + r, n = n0 + t*16 + ml)
  const int m0 = mBlk + wave * 16;
#pragma unroll
  for (int t = 0; t < 4; ++t) {
#pragma unroll
    for (int r = 0; r < 8; ++r) {
      const int m = m0 + kh * 8 + r;
      const int n = n0 + t * 16 + ml;
      const long long cidx = (long long)m * ldc + n + z * cSz;
      const float v = acc[t][r];
      if (MODE == 0) {
        ((float*)Cp)[cidx] = v;
      } else if (MODE == 1) {
        ((__bf16*)Cp)[cidx] = (__bf16)v;
      } else if (MODE == 2) {
        ((__bf16*)Cp)[cidx] = (__bf16)(v > 0.0f ? v : 0.0f);
      } else if (MODE == 3) {
        const float nv = residF[cidx] + v;
        residF[cidx]  = nv;
        residBf[cidx] = (__bf16)nv;
      } else if (MODE == 4) {
        ((float*)Cp)[cidx] = v + bias[n];
      }
    }
  }
}

// ---- Attention: single-pass flash loop (no max-subtraction in reference) ---
// grid = (T/16, B*H), block = 128 (4 waves). Block owns one 16-row Q tile;
// wave w owns V columns [w*64, w*64+64). Each wave redundantly computes S
// (cheap: K=D_K=32) and keeps its P tile in its own LDS slice.
__global__ __launch_bounds__(128) void attn_k(
    const __bf16* __restrict__ qb,   // [B, T, H*32]
    const __bf16* __restrict__ kb,   // [B, T, H*32]
    const __bf16* __restrict__ vt,   // [H, 256, B*T]  (V transposed)
    float* __restrict__ obuf)        // [B, H, T, 256]
{
  __shared__ __bf16 ptile[4][16][32];   // per-wave 16x32 P staging (4 KB)

  const int lane = threadIdx.x & 31;
  const int w    = threadIdx.x >> 5;
  const int ml   = lane & 15;
  const int kh   = lane >> 4;
  const int b    = blockIdx.y >> 3;
  const int hh   = blockIdx.y & 7;
  const int m0   = blockIdx.x * 16;

  // Q tile as A operand: one 16x16x32 WMMA covers the full D_K=32.
  const v16bf Qa =
      load_op(qb + (long long)(b * TT + m0 + ml) * D_MODEL + hh * D_K + kh * 8);

  v16bf ones;
#pragma unroll
  for (int i = 0; i < 16; ++i) ones[i] = (__bf16)1.0f;

  v8f acc[4];
#pragma unroll
  for (int t = 0; t < 4; ++t) acc[t] = zero8();
  v8f rs = zero8();   // running row sums (replicated across N lanes)

  for (int j0 = 0; j0 < m0 + 16; j0 += 32) {
    // S = Q x K^T for 32 key columns (two 16-col tiles). B-operand of K^T is
    // loaded exactly like an A row: lane n takes row (j0+n) of K.
    v16bf Kb0 = load_op(kb + (long long)(b * TT + j0 + ml) * D_MODEL + hh * D_K + kh * 8);
    v16bf Kb1 = load_op(kb + (long long)(b * TT + j0 + 16 + ml) * D_MODEL + hh * D_K + kh * 8);
    v8f S0 = wmma_bf16(Qa, Kb0, zero8());
    v8f S1 = wmma_bf16(Qa, Kb1, zero8());

    // Causal mask + exp, spill to LDS in A-operand layout (C->A transpose).
#pragma unroll
    for (int r = 0; r < 8; ++r) {
      const int m  = m0 + kh * 8 + r;
      const float p0 = (j0 + ml      <= m) ? __expf(S0[r]) : 0.0f;
      const float p1 = (j0 + 16 + ml <= m) ? __expf(S1[r]) : 0.0f;
      ptile[w][kh * 8 + r][ml]      = (__bf16)p0;
      ptile[w][kh * 8 + r][16 + ml] = (__bf16)p1;
    }
    // Same-wave DS RAW: DS ops are in-order per wave; compiler inserts waits.
    v16bf Pa = load_op(&ptile[w][ml][kh * 8]);

    rs = wmma_bf16(Pa, ones, rs);          // row sums via ones matrix
#pragma unroll
    for (int t = 0; t < 4; ++t) {
      const int nc = w * 64 + t * 16;
      v16bf Vb = load_op(vt + (long long)(hh * D_MODEL + nc + ml) * BT
                            + b * TT + j0 + kh * 8);
      acc[t] = wmma_bf16(Pa, Vb, acc[t]);
    }
  }

  // Normalize (rs is replicated across lanes 0..15 per row) and store.
#pragma unroll
  for (int t = 0; t < 4; ++t) {
#pragma unroll
    for (int r = 0; r < 8; ++r) {
      const int m  = m0 + kh * 8 + r;
      const int nc = w * 64 + t * 16 + ml;
      obuf[((long long)(b * N_HEAD + hh) * TT + m) * D_MODEL + nc] =
          acc[t][r] / (rs[r] + 1e-10f);
    }
  }
}

// ---- Elementwise kernels ---------------------------------------------------

// Transpose+convert: in[z][R][C] f32 -> out[z][C][R] bf16.
__global__ __launch_bounds__(256) void tcvt_k(const float* __restrict__ in,
                                              __bf16* __restrict__ out,
                                              int R, int C)
{
  const long long base = (long long)blockIdx.z * R * C;
  const int idx = blockIdx.x * 256 + threadIdx.x;
  if (idx < R * C) {
    const int r = idx / C, c = idx % C;
    out[base + (long long)c * R + r] = (__bf16)in[base + idx];
  }
}

__global__ __launch_bounds__(256) void embed_k(const int* __restrict__ x,
                                               const float* __restrict__ emb,
                                               float* __restrict__ h,
                                               __bf16* __restrict__ hbf)
{
  const int idx = blockIdx.x * 256 + threadIdx.x;   // over BT*D_MODEL
  const int tok = idx >> 8;
  const int c   = idx & 255;
  const float v = emb[(long long)x[tok] * D_MODEL + c];
  h[idx]   = v;
  hbf[idx] = (__bf16)v;
}

// h += sum over heads of obuf; refresh bf16 mirror. Deterministic.
__global__ __launch_bounds__(256) void attn_reduce_k(const float* __restrict__ obuf,
                                                     float* __restrict__ h,
                                                     __bf16* __restrict__ hbf)
{
  const int idx = blockIdx.x * 256 + threadIdx.x;   // over BT*D_MODEL
  const int bt = idx >> 8;
  const int c  = idx & 255;
  const int b  = bt >> 11;
  const int t  = bt & 2047;
  float s = h[idx];
#pragma unroll
  for (int hh = 0; hh < N_HEAD; ++hh)
    s += obuf[((long long)(b * N_HEAD + hh) * TT + t) * D_MODEL + c];
  h[idx]   = s;
  hbf[idx] = (__bf16)s;
}

// ---- Host side -------------------------------------------------------------

static char* carve(char*& p, size_t bytes) {
  char* r = p;
  p += (bytes + 255) & ~(size_t)255;
  return r;
}

extern "C" void kernel_launch(void* const* d_in, const int* in_sizes, int n_in,
                              void* d_out, int out_size, void* d_ws, size_t ws_size,
                              hipStream_t stream)
{
  const int*   x    = (const int*)d_in[0];
  const float* emb  = (const float*)d_in[1];
  const float* wq   = (const float*)d_in[2];   // [L,H,256,32]
  const float* wk   = (const float*)d_in[3];   // [L,H,256,32]
  const float* wv   = (const float*)d_in[4];   // [L,H,256,256]
  const float* mlp0 = (const float*)d_in[5];   // [L,256,1024]
  const float* mlp1 = (const float*)d_in[6];   // [L,1024,256]
  const float* ue   = (const float*)d_in[7];   // [256,32000]
  const float* bias = (const float*)d_in[8];   // [32000]
  float* out = (float*)d_out;
  (void)in_sizes; (void)n_in; (void)out_size; (void)ws_size;

  char* p = (char*)d_ws;
  float*  h    = (float*)carve(p,  (size_t)BT * D_MODEL * 4);
  __bf16* hbf  = (__bf16*)carve(p, (size_t)BT * D_MODEL * 2);
  __bf16* qb   = (__bf16*)carve(p, (size_t)BT * D_MODEL * 2);
  __bf16* kb   = (__bf16*)carve(p, (size_t)BT * D_MODEL * 2);
  __bf16* vtb  = (__bf16*)carve(p, (size_t)N_HEAD * D_MODEL * BT * 2);      // [H,256,BT]
  float*  obuf = (float*)carve(p,  (size_t)BB * N_HEAD * TT * D_MODEL * 4); // [B,H,T,256]
  __bf16* yb   = (__bf16*)carve(p, (size_t)BT * D_HIDDEN * 2);
  __bf16* wqT  = (__bf16*)carve(p, (size_t)N_LAYER * D_MODEL * D_MODEL * 2); // [L][H*32][256]
  __bf16* wkT  = (__bf16*)carve(p, (size_t)N_LAYER * D_MODEL * D_MODEL * 2);
  __bf16* wvT  = (__bf16*)carve(p, (size_t)N_LAYER * N_HEAD * D_MODEL * D_MODEL * 2);
  __bf16* m0T  = (__bf16*)carve(p, (size_t)N_LAYER * D_HIDDEN * D_MODEL * 2);
  __bf16* m1T  = (__bf16*)carve(p, (size_t)N_LAYER * D_MODEL * D_HIDDEN * 2);
  __bf16* ueT  = (__bf16*)carve(p, (size_t)N_DICT * D_MODEL * 2);

  // --- One-time weight transpose/convert (deterministic every call) ---
  tcvt_k<<<dim3((256 * 32 + 255) / 256, 1, N_LAYER * N_HEAD), 256, 0, stream>>>(wq, wqT, 256, 32);
  tcvt_k<<<dim3((256 * 32 + 255) / 256, 1, N_LAYER * N_HEAD), 256, 0, stream>>>(wk, wkT, 256, 32);
  tcvt_k<<<dim3((256 * 256 + 255) / 256, 1, N_LAYER * N_HEAD), 256, 0, stream>>>(wv, wvT, 256, 256);
  tcvt_k<<<dim3((256 * 1024 + 255) / 256, 1, N_LAYER), 256, 0, stream>>>(mlp0, m0T, 256, 1024);
  tcvt_k<<<dim3((1024 * 256 + 255) / 256, 1, N_LAYER), 256, 0, stream>>>(mlp1, m1T, 1024, 256);
  tcvt_k<<<dim3((256 * 32000 + 255) / 256, 1, 1), 256, 0, stream>>>(ue, ueT, 256, 32000);

  // --- Embedding gather ---
  embed_k<<<dim3(BT * D_MODEL / 256), 256, 0, stream>>>(x, emb, h, hbf);

  for (int l = 0; l < N_LAYER; ++l) {
    const long long wqkOff = (long long)l * D_MODEL * D_MODEL;
    // Q,K: [4096,256] = hbf[4096,256] x wT[256,256]^T  (bf16 out, head-major cols)
    gemm_k<1><<<dim3(D_MODEL / 64, BT / 128, 1), 256, 0, stream>>>(
        hbf, D_MODEL, 0, wqT + wqkOff, D_MODEL, 0, qb, D_MODEL, 0, D_MODEL,
        nullptr, nullptr, nullptr);
    gemm_k<1><<<dim3(D_MODEL / 64, BT / 128, 1), 256, 0, stream>>>(
        hbf, D_MODEL, 0, wkT + wqkOff, D_MODEL, 0, kb, D_MODEL, 0, D_MODEL,
        nullptr, nullptr, nullptr);
    // V (transposed): vt[h][n][t] = sum_m wvT[l,h][n][m] * hbf[t][m]
    gemm_k<1><<<dim3(BT / 64, D_MODEL / 128, N_HEAD), 256, 0, stream>>>(
        wvT + (long long)l * N_HEAD * D_MODEL * D_MODEL, D_MODEL,
        (long long)D_MODEL * D_MODEL,
        hbf, D_MODEL, 0,
        vtb, BT, (long long)D_MODEL * BT, D_MODEL,
        nullptr, nullptr, nullptr);

    // Flash attention per (b,h), 16-row Q tiles.
    attn_k<<<dim3(TT / 16, BB * N_HEAD), 128, 0, stream>>>(qb, kb, vtb, obuf);
    attn_reduce_k<<<dim3(BT * D_MODEL / 256), 256, 0, stream>>>(obuf, h, hbf);

    // MLP0: y = relu(hbf x mlp0) -> bf16
    gemm_k<2><<<dim3(D_HIDDEN / 64, BT / 128, 1), 256, 0, stream>>>(
        hbf, D_MODEL, 0, m0T + (long long)l * D_HIDDEN * D_MODEL, D_MODEL, 0,
        yb, D_HIDDEN, 0, D_MODEL, nullptr, nullptr, nullptr);
    // MLP1: h += y x mlp1 (residual, refresh bf16 mirror)
    gemm_k<3><<<dim3(D_MODEL / 64, BT / 128, 1), 256, 0, stream>>>(
        yb, D_HIDDEN, 0, m1T + (long long)l * D_MODEL * D_HIDDEN, D_HIDDEN, 0,
        h, D_MODEL, 0, D_HIDDEN, h, hbf, nullptr);
  }

  // Unembedding + bias -> f32 logits [4096, 32000]
  gemm_k<4><<<dim3(N_DICT / 64, BT / 128, 1), 256, 0, stream>>>(
      hbf, D_MODEL, 0, ueT, D_MODEL, 0, out, N_DICT, 0, D_MODEL,
      nullptr, nullptr, bias);
}
